// MultiQueryAttention_67748814127501
// MI455X (gfx1250) — compile-verified, matrix operand order checked by reading, not run on hardware
//
#include <hip/hip_runtime.h>

#define D_IN 2048
#define N_HEADS 16
#define HEAD_DIM 128
#define SEQ 2048
#define D_OUT (N_HEADS * HEAD_DIM)

typedef __attribute__((ext_vector_type(16))) __bf16 v16bf;
typedef __attribute__((ext_vector_type(8)))  float  v8f;
typedef __attribute__((ext_vector_type(4)))  unsigned int v4u;
typedef __attribute__((ext_vector_type(8)))  int    v8i;
typedef __attribute__((ext_vector_type(4)))  int    v4i;

// Load two contiguous 8-element bf16 runs (16B each) into WMMA A/B operand order.
__device__ __forceinline__ v16bf ld_pair16(const __bf16* p0, const __bf16* p1) {
  union { uint4 u[2]; v16bf v; } r;
  r.u[0] = *reinterpret_cast<const uint4*>(p0);
  r.u[1] = *reinterpret_cast<const uint4*>(p1);
  return r.v;
}

__device__ __forceinline__ v8f wmma_bf16(v16bf a, v16bf b, v8f c) {
  return __builtin_amdgcn_wmma_f32_16x16x32_bf16(false, a, false, b, (short)0, c,
                                                 false, false);
}

// TDM: DMA a 2D tile (rows16 rows x 64B) from global (row stride = strideElems4B x 4B)
// into LDS at byte offset ldsOff.  D# built per CDNA5 ISA ch.8 (2D form: groups
// 2/3 zero => tile_dim2/3 unused).  data_size = 4B.
__device__ __forceinline__ void tdm_load_tile_64x64B(const void* gptr, unsigned ldsOff,
                                                     unsigned rows, unsigned rowElems4B,
                                                     unsigned strideElems4B) {
  unsigned long long ga = (unsigned long long)(uintptr_t)gptr;
  v4u g0 = { 1u,                                  // count=1, user descriptor
             ldsOff,                              // lds_addr
             (unsigned)ga,                        // global_addr[31:0]
             (unsigned)((ga >> 32) & 0x01FFFFFFu) | (2u << 30) };  // addr[56:32] | type=2
  const unsigned td0 = 0x100000u, td1 = 0x100000u;   // huge tensor dims: OOB never hit
  v8i g1 = { (int)(2u << 16),                                    // data_size=4B
             (int)((td0 & 0xFFFFu) << 16),                       // tensor_dim0 lo
             (int)(((td0 >> 16) & 0xFFFFu) | ((td1 & 0xFFFFu) << 16)),
             (int)(((td1 >> 16) & 0xFFFFu) | (rowElems4B << 16)),// tile_dim0
             (int)rows,                                          // tile_dim1 (tile_dim2=0)
             (int)strideElems4B,                                 // tensor_dim0_stride[31:0]
             0, 0 };
  v4i gz = { 0, 0, 0, 0 };
#if __clang_major__ >= 23
  v8i gz8 = { 0, 0, 0, 0, 0, 0, 0, 0 };
  __builtin_amdgcn_tensor_load_to_lds(g0, g1, gz, gz, gz8, 0);
#else
  __builtin_amdgcn_tensor_load_to_lds(g0, g1, gz, gz, 0);
#endif
}

// ---------------------------------------------------------------- cast f32->bf16
__global__ void cast_f32_bf16_k(const float* __restrict__ in, __bf16* __restrict__ out,
                                int n) {
  for (int i = blockIdx.x * blockDim.x + threadIdx.x; i < n;
       i += gridDim.x * blockDim.x)
    out[i] = (__bf16)in[i];
}

// --------------------------------------------------- C[M,N](f32) = A[M,K] * W[N,K]^T
// Direct-load variant (small N: K/V projections). One wave computes a 16x64 tile.
__global__ void gemm_bf16_k(const __bf16* __restrict__ A, const __bf16* __restrict__ W,
                            float* __restrict__ C, int K, int ldc) {
  const int lane = threadIdx.x & 31;
  const int r    = lane & 15;
  const int off  = (lane < 16) ? 0 : 8;
  const int m0   = blockIdx.x * 16;
  const int n0   = blockIdx.y * 64;

  const __bf16* arow  = A + (long)(m0 + r) * K;
  const __bf16* wrow0 = W + (long)(n0 + r) * K;

  v8f acc[4] = {};
  for (int k = 0; k < K; k += 32) {
    v16bf a = ld_pair16(arow + k + off, arow + k + 16 + off);
#pragma unroll
    for (int t = 0; t < 4; ++t) {
      const __bf16* wrow = wrow0 + (long)(t * 16) * K;
      v16bf b = ld_pair16(wrow + k + off, wrow + k + 16 + off);
      acc[t] = wmma_bf16(a, b, acc[t]);
    }
  }
#pragma unroll
  for (int t = 0; t < 4; ++t) {
#pragma unroll
    for (int rr = 0; rr < 8; ++rr) {
      int row = m0 + rr + off;
      int col = n0 + t * 16 + r;
      C[(long)row * ldc + col] = acc[t][rr];
    }
  }
}

// --------------------------------------------------- TDM+LDS variant (big GEMMs).
// 8 waves/WG share a 64-row W tile staged by the Tensor Data Mover, double-buffered.
// WG tile = 128(M) x 64(N); wave tile = 16 x 64.
__global__ void gemm_bf16_tdm_k(const __bf16* __restrict__ A, const __bf16* __restrict__ W,
                                float* __restrict__ C, int K, int ldc) {
  extern __shared__ __bf16 smem[];            // 2 buffers x (64 x 32) bf16 = 8 KB
  const int wave = threadIdx.x >> 5;
  const int lane = threadIdx.x & 31;
  const int r    = lane & 15;
  const int off  = (lane < 16) ? 0 : 8;
  const int m0   = blockIdx.x * 128 + wave * 16;
  const int n0   = blockIdx.y * 64;

  const __bf16* arow = A + (long)(m0 + r) * K;
  const int nch = K >> 5;                      // 32-wide K chunks

  if (wave == 0)
    tdm_load_tile_64x64B(W + (long)n0 * K, 0u, 64u, 16u, (unsigned)(K >> 1));

  v8f acc[4] = {};
  for (int i = 0; i < nch; ++i) {
    if (wave == 0) {
      if (i + 1 < nch) {
        tdm_load_tile_64x64B(W + (long)n0 * K + (long)(i + 1) * 32,
                             (unsigned)(((i + 1) & 1) * 4096), 64u, 16u,
                             (unsigned)(K >> 1));
        __builtin_amdgcn_s_wait_tensorcnt(1);  // in-order: chunk i has landed
      } else {
        __builtin_amdgcn_s_wait_tensorcnt(0);
      }
    }
    __syncthreads();                           // tile i visible to all waves
    const int k = i << 5;
    v16bf a = ld_pair16(arow + k + off, arow + k + 16 + off);
    const __bf16* wt = smem + (size_t)((i & 1) * 2048);
#pragma unroll
    for (int t = 0; t < 4; ++t) {
      const __bf16* wrow = wt + (t * 16 + r) * 32;
      v16bf b = ld_pair16(wrow + off, wrow + 16 + off);   // ds_load_b128 x2
      acc[t] = wmma_bf16(a, b, acc[t]);
    }
    __syncthreads();                           // all consumed before buffer reuse
  }
#pragma unroll
  for (int t = 0; t < 4; ++t) {
#pragma unroll
    for (int rr = 0; rr < 8; ++rr) {
      int row = m0 + rr + off;
      int col = n0 + t * 16 + r;
      C[(long)row * ldc + col] = acc[t][rr];
    }
  }
}

// --------------------------------------- fused RMSNorm + RoPE, one wave per (s,head)
__global__ void rmsnorm_rope_k(const float* __restrict__ raw, __bf16* __restrict__ outp,
                               const float* __restrict__ scale,
                               const float* __restrict__ shift,
                               const float* __restrict__ sinT,
                               const float* __restrict__ cosT, int nheads) {
  const int s = blockIdx.x, h = blockIdx.y, lane = threadIdx.x;
  const float* x = raw + ((long)s * nheads + h) * HEAD_DIM;
  float v[4], n[4];
  float ss = 0.f;
#pragma unroll
  for (int i = 0; i < 4; ++i) { v[i] = x[lane + 32 * i]; ss += v[i] * v[i]; }
#pragma unroll
  for (int o = 16; o >= 1; o >>= 1) ss += __shfl_xor(ss, o, 32);
  const float rinv = rsqrtf(ss * (1.0f / HEAD_DIM) + 1e-6f);
#pragma unroll
  for (int i = 0; i < 4; ++i) {
    int d = lane + 32 * i;
    n[i] = v[i] * rinv * (1.f + scale[d]) + shift[d];
  }
  __bf16* op = outp + ((long)s * nheads + h) * HEAD_DIM;
#pragma unroll
  for (int i = 0; i < 4; ++i) {
    int d = lane + 32 * i;
    float rot = (i < 2) ? -n[i + 2] : n[i - 2];   // partner d±64 stays in-lane
    op[d] = (__bf16)(n[i] * cosT[s * HEAD_DIM + d] + rot * sinT[s * HEAD_DIM + d]);
  }
}

// ------------------------------------------------ V [s][d] f32 -> V^T [d][s] bf16
__global__ void vtrans_k(const float* __restrict__ vraw, __bf16* __restrict__ vT) {
  int i = blockIdx.x * blockDim.x + threadIdx.x;
  if (i < SEQ * HEAD_DIM) {
    int s = i >> 7, d = i & 127;
    vT[(long)d * SEQ + s] = (__bf16)vraw[i];
  }
}

// ---------------------------------------------------------------- flash attention
// One wave owns 16 query rows of one head. Computes S^T = K*Q^T so softmax rows are
// per-lane (one shfl_xor(16) merge) and P^T is already in B-operand layout for
// O^T = V^T * P^T.  32 keys / iteration, causal.
__global__ void mqa_attn_k(const __bf16* __restrict__ qn, const __bf16* __restrict__ kn,
                           const __bf16* __restrict__ vT, __bf16* __restrict__ attn) {
  const int head  = blockIdx.y;
  const int wave  = threadIdx.x >> 5;
  const int lane  = threadIdx.x & 31;
  const int r     = lane & 15;
  const int off   = (lane < 16) ? 0 : 8;
  const int qbase = blockIdx.x * 64 + wave * 16;
  const float scl = 0.08838834764831845f;   // HEAD_DIM^-0.5

  const __bf16* qrow = qn + (long)(qbase + r) * D_OUT + head * HEAD_DIM;
  v16bf qb[4];
#pragma unroll
  for (int c = 0; c < 4; ++c)
    qb[c] = ld_pair16(qrow + c * 32 + off, qrow + c * 32 + 16 + off);

  v8f oacc[8] = {};                         // O^T: 8 d-tiles x (16d x 16q)
  float mrun = -1e30f, lrun = 0.f;
  const int qlim = qbase + 15;

  for (int j = 0; j <= qlim; j += 32) {
    v8f s0 = {}, s1 = {};
    const __bf16* krow0 = kn + (long)(j + r) * HEAD_DIM;
    const __bf16* krow1 = krow0 + 16 * HEAD_DIM;
#pragma unroll
    for (int c = 0; c < 4; ++c) {
      v16bf a0 = ld_pair16(krow0 + c * 32 + off, krow0 + c * 32 + 16 + off);
      s0 = wmma_bf16(a0, qb[c], s0);
      v16bf a1 = ld_pair16(krow1 + c * 32 + off, krow1 + c * 32 + 16 + off);
      s1 = wmma_bf16(a1, qb[c], s1);
    }
    const int qg = qbase + r;
    float sv0[8], sv1[8];
#pragma unroll
    for (int rr = 0; rr < 8; ++rr) {
      int k0 = j + rr + off;
      sv0[rr] = (k0      <= qg) ? s0[rr] * scl : -1e30f;
      sv1[rr] = (k0 + 16 <= qg) ? s1[rr] * scl : -1e30f;
    }
    float mloc = -1e30f;
#pragma unroll
    for (int rr = 0; rr < 8; ++rr) mloc = fmaxf(mloc, fmaxf(sv0[rr], sv1[rr]));
    mloc = fmaxf(mloc, __shfl_xor(mloc, 16, 32));
    float mnew  = fmaxf(mrun, mloc);
    float alpha = __expf(mrun - mnew);
    float lloc  = 0.f;
    v16bf pb;                               // P^T directly in B-operand layout
#pragma unroll
    for (int rr = 0; rr < 8; ++rr) {
      float p0 = __expf(sv0[rr] - mnew);
      float p1 = __expf(sv1[rr] - mnew);
      lloc += p0 + p1;
      pb[rr]     = (__bf16)p0;
      pb[rr + 8] = (__bf16)p1;
    }
    lloc += __shfl_xor(lloc, 16, 32);
    lrun = lrun * alpha + lloc;
    mrun = mnew;
#pragma unroll
    for (int t = 0; t < 8; ++t)
#pragma unroll
      for (int rr = 0; rr < 8; ++rr) oacc[t][rr] *= alpha;
#pragma unroll
    for (int t = 0; t < 8; ++t) {
      const __bf16* vrow = vT + (long)(t * 16 + r) * SEQ;
      v16bf va = ld_pair16(vrow + j + off, vrow + j + 16 + off);
      oacc[t] = wmma_bf16(va, pb, oacc[t]);
    }
  }

  const float inv = 1.f / lrun;
  __bf16* orow = attn + (long)(qbase + r) * D_OUT + head * HEAD_DIM;
#pragma unroll
  for (int t = 0; t < 8; ++t) {
#pragma unroll
    for (int rr = 0; rr < 8; rr += 2) {
      union { __bf16 h[2]; unsigned u; } pk;
      pk.h[0] = (__bf16)(oacc[t][rr]     * inv);
      pk.h[1] = (__bf16)(oacc[t][rr + 1] * inv);
      *reinterpret_cast<unsigned*>(orow + t * 16 + off + rr) = pk.u;
    }
  }
}

// =================================================================== launcher
extern "C" void kernel_launch(void* const* d_in, const int* in_sizes, int n_in,
                              void* d_out, int out_size, void* d_ws, size_t ws_size,
                              hipStream_t stream) {
  (void)in_sizes; (void)n_in; (void)out_size; (void)ws_size;
  const float* x      = (const float*)d_in[0];
  // d_in[1] = mask (bool): causal, recomputed analytically in-kernel
  const float* sinT   = (const float*)d_in[2];
  const float* cosT   = (const float*)d_in[3];
  const float* w_q    = (const float*)d_in[4];
  const float* w_k    = (const float*)d_in[5];
  const float* w_v    = (const float*)d_in[6];
  const float* w_o    = (const float*)d_in[7];
  const float* q_scale = (const float*)d_in[8];
  const float* q_shift = (const float*)d_in[9];
  const float* k_scale = (const float*)d_in[10];
  const float* k_shift = (const float*)d_in[11];
  float* outp = (float*)d_out;

  char* w = (char*)d_ws;
  auto alloc = [&](size_t bytes) {
    void* p = (void*)w;
    w += (bytes + 255) & ~(size_t)255;
    return p;
  };
  __bf16* xb   = (__bf16*)alloc((size_t)SEQ * D_IN * 2);
  __bf16* wqb  = (__bf16*)alloc((size_t)D_OUT * D_IN * 2);
  __bf16* wkb  = (__bf16*)alloc((size_t)HEAD_DIM * D_IN * 2);
  __bf16* wvb  = (__bf16*)alloc((size_t)HEAD_DIM * D_IN * 2);
  __bf16* wob  = (__bf16*)alloc((size_t)D_IN * D_OUT * 2);
  float*  qraw = (float*) alloc((size_t)SEQ * D_OUT * 4);
  float*  kraw = (float*) alloc((size_t)SEQ * HEAD_DIM * 4);
  float*  vraw = (float*) alloc((size_t)SEQ * HEAD_DIM * 4);
  __bf16* qnb  = (__bf16*)alloc((size_t)SEQ * D_OUT * 2);
  __bf16* knb  = (__bf16*)alloc((size_t)SEQ * HEAD_DIM * 2);
  __bf16* vTb  = (__bf16*)alloc((size_t)HEAD_DIM * SEQ * 2);
  __bf16* attn = (__bf16*)alloc((size_t)SEQ * D_OUT * 2);

  // 1) downcast activations + weights to bf16 (everything stays L2-resident)
  cast_f32_bf16_k<<<dim3(1024), dim3(256), 0, stream>>>(x,   xb,  SEQ * D_IN);
  cast_f32_bf16_k<<<dim3(1024), dim3(256), 0, stream>>>(w_q, wqb, D_OUT * D_IN);
  cast_f32_bf16_k<<<dim3(256),  dim3(256), 0, stream>>>(w_k, wkb, HEAD_DIM * D_IN);
  cast_f32_bf16_k<<<dim3(256),  dim3(256), 0, stream>>>(w_v, wvb, HEAD_DIM * D_IN);
  cast_f32_bf16_k<<<dim3(1024), dim3(256), 0, stream>>>(w_o, wob, D_IN * D_OUT);

  // 2) projections: big GEMMs use TDM double-buffered LDS staging of the W tile
  gemm_bf16_tdm_k<<<dim3(SEQ / 128, D_OUT / 64), dim3(256), 8192, stream>>>(xb, wqb, qraw, D_IN, D_OUT);
  gemm_bf16_k<<<dim3(SEQ / 16, HEAD_DIM / 64), dim3(32), 0, stream>>>(xb, wkb, kraw, D_IN, HEAD_DIM);
  gemm_bf16_k<<<dim3(SEQ / 16, HEAD_DIM / 64), dim3(32), 0, stream>>>(xb, wvb, vraw, D_IN, HEAD_DIM);

  // 3) RMSNorm + RoPE (q per head, k single KV head); V transposed once (MQA)
  rmsnorm_rope_k<<<dim3(SEQ, N_HEADS), dim3(32), 0, stream>>>(qraw, qnb, q_scale, q_shift, sinT, cosT, N_HEADS);
  rmsnorm_rope_k<<<dim3(SEQ, 1),       dim3(32), 0, stream>>>(kraw, knb, k_scale, k_shift, sinT, cosT, 1);
  vtrans_k<<<dim3(SEQ * HEAD_DIM / 256), dim3(256), 0, stream>>>(vraw, vTb);

  // 4) causal flash attention, all-WMMA
  mqa_attn_k<<<dim3(SEQ / 64, N_HEADS), dim3(128), 0, stream>>>(qnb, knb, vTb, attn);

  // 5) output projection -> f32 d_out (TDM variant)
  gemm_bf16_tdm_k<<<dim3(SEQ / 128, D_IN / 64), dim3(256), 8192, stream>>>(attn, wob, outp, D_OUT, D_IN);
}
